// GCNNet_35725537968345
// MI455X (gfx1250) — matrix operand order repaired and checked by reading, not compiled, not run on hardware
//
#include <hip/hip_runtime.h>

typedef __bf16 bf16_t;
typedef __attribute__((ext_vector_type(16))) __bf16 v16bf;
typedef __attribute__((ext_vector_type(8)))  float  v8f;

#define HDIM 128

// ---------------- utility kernels ----------------

// Pack conv weights into WMMA B-fragment-major bf16 layout.
// Packed index: (((l*32 + (j*4+ks))*32 + lane)*16 + e)
//   maps to source W_l[k][c] with k = ks*32 + (lane>>4)*16 + e, c = j*16 + (lane&15)
__global__ __launch_bounds__(256) void k_pack_w(const float* __restrict__ w,
                                                bf16_t* __restrict__ wpk, int n) {
  int i = blockIdx.x * 256 + threadIdx.x;
  if (i >= n) return;
  int e    = i & 15;
  int lane = (i >> 4) & 31;
  int f    = (i >> 9) & 31;
  int l    = i >> 14;
  int ks   = f & 3;
  int j    = f >> 2;
  int half = lane >> 4;
  int mr   = lane & 15;
  int k = ks * 32 + half * 16 + e;
  int c = j * 16 + mr;
  wpk[i] = (bf16_t)w[l * HDIM * HDIM + k * HDIM + c];
}

__global__ __launch_bounds__(256) void k_fill(float* __restrict__ p, float v, int n) {
  int i = blockIdx.x * 256 + threadIdx.x;
  if (i < n) p[i] = v;
}

__global__ __launch_bounds__(256) void k_count_deg(const int* __restrict__ dst,
                                                   float* __restrict__ deg,
                                                   int etot, int e) {
  int i = blockIdx.x * 256 + threadIdx.x;
  if (i >= etot) return;
  int d = (i < e) ? dst[i] : (i - e);           // self-loop tail
  __hip_atomic_fetch_add(&deg[d], 1.0f, __ATOMIC_RELAXED, __HIP_MEMORY_SCOPE_AGENT);
}

__global__ __launch_bounds__(256) void k_dis(const float* __restrict__ deg,
                                             float* __restrict__ dis, int n) {
  int i = blockIdx.x * 256 + threadIdx.x;
  if (i < n) {
    float d = deg[i];
    dis[i] = (d > 0.0f) ? rsqrtf(d) : 0.0f;
  }
}

__global__ __launch_bounds__(256) void k_init_bias(float* __restrict__ acc,
                                                   const float* __restrict__ b, int n) {
  int i = blockIdx.x * 256 + threadIdx.x;
  if (i < n) acc[i] = b[i & (HDIM - 1)];
}

__global__ __launch_bounds__(256) void k_relu(const float* __restrict__ a,
                                              float* __restrict__ h, int n) {
  int i = blockIdx.x * 256 + threadIdx.x;
  if (i < n) h[i] = fmaxf(a[i], 0.0f);
}

// ---------------- bf16 WMMA GEMM: Out[rows x 128] = A[rows x 128] * W[128 x 128] ----------------
// 8 waves/block, 16 rows/wave. W is pre-swizzled fragment-major bf16, staged into LDS
// via the gfx1250 async global->LDS data path (no VGPR round trip); each B fragment is
// one contiguous 32-byte LDS read per lane (2x ds_load_b128).
// FULL=true blocks have no store guards; the single tail block uses FULL=false.

template <bool FULL>
__global__ __launch_bounds__(256) void k_gemm_bf16(const float* __restrict__ A,
                                                   const bf16_t* __restrict__ Wpk,
                                                   float* __restrict__ Out,
                                                   int rows, int blockBase) {
  __shared__ bf16_t Ws[HDIM * HDIM];   // 32 KB, fragment-major; sole LDS object -> base offset 0
  int t = threadIdx.x;
  {
    // Each thread async-copies 8 x 16B (stride 4 KB); IOFFSET is added to BOTH the
    // LDS and global addresses, so one register pair serves all eight transfers.
    unsigned lds = (unsigned)t * 16u;
    const bf16_t* g = Wpk + t * 8;
    asm volatile(
        "global_load_async_to_lds_b128 %0, %1, off\n\t"
        "global_load_async_to_lds_b128 %0, %1, off offset:4096\n\t"
        "global_load_async_to_lds_b128 %0, %1, off offset:8192\n\t"
        "global_load_async_to_lds_b128 %0, %1, off offset:12288\n\t"
        "global_load_async_to_lds_b128 %0, %1, off offset:16384\n\t"
        "global_load_async_to_lds_b128 %0, %1, off offset:20480\n\t"
        "global_load_async_to_lds_b128 %0, %1, off offset:24576\n\t"
        "global_load_async_to_lds_b128 %0, %1, off offset:28672\n\t"
        "s_wait_asynccnt 0"
        :
        : "v"(lds), "v"(g)
        : "memory");
  }
  __syncthreads();

  const int wave = t >> 5;
  const int lane = t & 31;
  const int half = lane >> 4;
  const int mr   = lane & 15;
  const int rowBase = (blockBase + (int)blockIdx.x) * 128 + wave * 16;

  // ---- A fragments: 16 rows x 128 K, 4 K-steps of 32 ----
  int m  = rowBase + mr;
  int mc = (FULL || m < rows) ? m : (rows - 1);  // clamp so EXEC stays full at WMMA
  const float4* arow4 = (const float4*)(A + (size_t)mc * HDIM);

  v16bf afrag[4];
#pragma unroll
  for (int ks = 0; ks < 4; ++ks) {
    // lane's K range: [ks*32 + half*16, +16)
    int b4 = ks * 8 + half * 4;                  // float4 index
    float4 a0 = arow4[b4 + 0];
    float4 a1 = arow4[b4 + 1];
    float4 a2 = arow4[b4 + 2];
    float4 a3 = arow4[b4 + 3];
    afrag[ks][0]  = (bf16_t)a0.x;  afrag[ks][1]  = (bf16_t)a0.y;
    afrag[ks][2]  = (bf16_t)a0.z;  afrag[ks][3]  = (bf16_t)a0.w;
    afrag[ks][4]  = (bf16_t)a1.x;  afrag[ks][5]  = (bf16_t)a1.y;
    afrag[ks][6]  = (bf16_t)a1.z;  afrag[ks][7]  = (bf16_t)a1.w;
    afrag[ks][8]  = (bf16_t)a2.x;  afrag[ks][9]  = (bf16_t)a2.y;
    afrag[ks][10] = (bf16_t)a2.z;  afrag[ks][11] = (bf16_t)a2.w;
    afrag[ks][12] = (bf16_t)a3.x;  afrag[ks][13] = (bf16_t)a3.y;
    afrag[ks][14] = (bf16_t)a3.z;  afrag[ks][15] = (bf16_t)a3.w;
  }

  const v16bf* Wfrag = (const v16bf*)Ws;         // 32 B per lane per fragment

  // single base address; all 64 stores use immediate offsets (j*64 + i*512 bytes)
  float* pbase = Out + (size_t)(rowBase + half * 8) * HDIM + mr;

  // ---- 8 column tiles x 4 K-step WMMA chains ----
#pragma unroll
  for (int j = 0; j < 8; ++j) {
    v8f acc = {};
#pragma unroll
    for (int ks = 0; ks < 4; ++ks) {
      v16bf bfrag = Wfrag[(j * 4 + ks) * 32 + lane];   // 2x ds_load_b128
      acc = __builtin_amdgcn_wmma_f32_16x16x32_bf16(
          false, afrag[ks], false, bfrag, (short)0, acc, false, false);
    }
#pragma unroll
    for (int i = 0; i < 8; ++i) {
      if (FULL || (rowBase + half * 8 + i < rows))
        pbase[j * 16 + i * HDIM] = acc[i];       // C/D: VGPR i -> M=i (+8 for hi lanes)
    }
  }
}

// ---------------- edge scatter: one wave per edge, float4 per lane ----------------

__global__ __launch_bounds__(256) void k_scatter(const float* __restrict__ hw,
                                                 const int* __restrict__ srcA,
                                                 const int* __restrict__ dstA,
                                                 const float* __restrict__ dis,
                                                 float* __restrict__ acc,
                                                 int etot, int e) {
  int w = (int)((blockIdx.x * 256u + threadIdx.x) >> 5);
  int lane = threadIdx.x & 31;
  if (w >= etot) return;
  int s, d;
  if (w < e) { s = srcA[w]; d = dstA[w]; }
  else       { s = w - e;  d = s; }              // self-loop
  float nrm = dis[s] * dis[d];
  float4 v = ((const float4*)(hw + (size_t)s * HDIM))[lane];
  float* o = acc + (size_t)d * HDIM + lane * 4;
  __hip_atomic_fetch_add(o + 0, v.x * nrm, __ATOMIC_RELAXED, __HIP_MEMORY_SCOPE_AGENT);
  __hip_atomic_fetch_add(o + 1, v.y * nrm, __ATOMIC_RELAXED, __HIP_MEMORY_SCOPE_AGENT);
  __hip_atomic_fetch_add(o + 2, v.z * nrm, __ATOMIC_RELAXED, __HIP_MEMORY_SCOPE_AGENT);
  __hip_atomic_fetch_add(o + 3, v.w * nrm, __ATOMIC_RELAXED, __HIP_MEMORY_SCOPE_AGENT);
}

// ---------------- mean pool ----------------

__global__ __launch_bounds__(256) void k_pool(const float* __restrict__ h,
                                              const int* __restrict__ batch,
                                              float* __restrict__ pooled,
                                              float* __restrict__ cnt, int n) {
  int w = (int)((blockIdx.x * 256u + threadIdx.x) >> 5);
  int lane = threadIdx.x & 31;
  if (w >= n) return;
  int b = batch[w];
  float4 v = ((const float4*)(h + (size_t)w * HDIM))[lane];
  float* o = pooled + (size_t)b * HDIM + lane * 4;
  __hip_atomic_fetch_add(o + 0, v.x, __ATOMIC_RELAXED, __HIP_MEMORY_SCOPE_AGENT);
  __hip_atomic_fetch_add(o + 1, v.y, __ATOMIC_RELAXED, __HIP_MEMORY_SCOPE_AGENT);
  __hip_atomic_fetch_add(o + 2, v.z, __ATOMIC_RELAXED, __HIP_MEMORY_SCOPE_AGENT);
  __hip_atomic_fetch_add(o + 3, v.w, __ATOMIC_RELAXED, __HIP_MEMORY_SCOPE_AGENT);
  if (lane == 0)
    __hip_atomic_fetch_add(&cnt[b], 1.0f, __ATOMIC_RELAXED, __HIP_MEMORY_SCOPE_AGENT);
}

__global__ __launch_bounds__(256) void k_pool_div(float* __restrict__ pooled,
                                                  const float* __restrict__ cnt, int n) {
  int i = blockIdx.x * 256 + threadIdx.x;
  if (i < n) pooled[i] /= fmaxf(cnt[i >> 7], 1.0f);
}

// ---------------- MLP head ----------------

__global__ __launch_bounds__(256) void k_lin1(const float* __restrict__ p,
                                              const float* __restrict__ w1,
                                              const float* __restrict__ b1,
                                              float* __restrict__ z) {
  int i = blockIdx.x * 256 + threadIdx.x;        // < 128*128
  int g = i >> 7, n = i & 127;
  float s = b1[n];
  for (int k = 0; k < HDIM; ++k) s += p[g * HDIM + k] * w1[k * HDIM + n];
  z[i] = fmaxf(s, 0.0f);
}

__global__ __launch_bounds__(128) void k_head(const float* __restrict__ z,
                                              const float* __restrict__ w2,
                                              const float* __restrict__ b2,
                                              float* __restrict__ out) {
  int g = threadIdx.x;
  if (g >= 128) return;
  float lg[10];
  float mx = -1e30f;
  for (int c = 0; c < 10; ++c) {
    float s = b2[c];
    for (int k = 0; k < HDIM; ++k) s += z[g * HDIM + k] * w2[k * 10 + c];
    lg[c] = s;
    mx = fmaxf(mx, s);
  }
  float se = 0.0f;
  for (int c = 0; c < 10; ++c) se += __expf(lg[c] - mx);
  float lse = __logf(se);
  for (int c = 0; c < 10; ++c) out[g * 10 + c] = lg[c] - mx - lse;
}

// ---------------- launch ----------------

extern "C" void kernel_launch(void* const* d_in, const int* in_sizes, int n_in,
                              void* d_out, int out_size, void* d_ws, size_t ws_size,
                              hipStream_t stream) {
  const int N = 50000, E = 625000, L = 4, G = 128;
  const int ETOT = E + N;

  const float* x       = (const float*)d_in[0];
  const int*   ei      = (const int*)d_in[1];   // [2, E]
  const int*   batch   = (const int*)d_in[2];
  const float* conv_w  = (const float*)d_in[3]; // [L,128,128]
  const float* conv_b  = (const float*)d_in[4]; // [L,128]
  const float* w1      = (const float*)d_in[5];
  const float* b1      = (const float*)d_in[6];
  const float* w2      = (const float*)d_in[7];
  const float* b2      = (const float*)d_in[8];
  float* out = (float*)d_out;

  char* ws = (char*)d_ws;
  size_t off = 0;
  auto alloc = [&](size_t bytes) -> void* {
    void* p = ws + off;
    off = (off + bytes + 255) & ~(size_t)255;
    return p;
  };
  bf16_t* Wpk   = (bf16_t*)alloc((size_t)L * HDIM * HDIM * sizeof(bf16_t));
  float* hw     = (float*)alloc((size_t)N * HDIM * sizeof(float));
  float* acc    = (float*)alloc((size_t)N * HDIM * sizeof(float));
  float* hbuf   = (float*)alloc((size_t)N * HDIM * sizeof(float));
  float* deg    = (float*)alloc((size_t)N * sizeof(float));
  float* dis    = (float*)alloc((size_t)N * sizeof(float));
  float* pooled = (float*)alloc((size_t)G * HDIM * sizeof(float));
  float* cnt    = (float*)alloc((size_t)G * sizeof(float));
  float* z      = (float*)alloc((size_t)G * HDIM * sizeof(float));

  const int* srcA = ei;
  const int* dstA = ei + E;

  // weights -> packed bf16 fragments once per call
  k_pack_w<<<(L * HDIM * HDIM + 255) / 256, 256, 0, stream>>>(conv_w, Wpk, L * HDIM * HDIM);

  // degrees -> dis = rsqrt(deg)
  k_fill<<<(N + 255) / 256, 256, 0, stream>>>(deg, 0.0f, N);
  k_count_deg<<<(ETOT + 255) / 256, 256, 0, stream>>>(dstA, deg, ETOT, E);
  k_dis<<<(N + 255) / 256, 256, 0, stream>>>(deg, dis, N);

  const int fullBlocks = N / 128;                // 390 guard-free blocks
  const bool hasTail   = (N % 128) != 0;         // 80-row tail block
  const int nf = N * HDIM;
  const int scatterBlocks = (int)(((size_t)ETOT * 32 + 255) / 256);

  for (int l = 0; l < L; ++l) {
    const float* hin = (l == 0) ? x : hbuf;
    const bf16_t* wl = Wpk + (size_t)l * HDIM * HDIM;
    k_gemm_bf16<true><<<fullBlocks, 256, 0, stream>>>(hin, wl, hw, N, 0);
    if (hasTail)
      k_gemm_bf16<false><<<1, 256, 0, stream>>>(hin, wl, hw, N, fullBlocks);
    k_init_bias<<<(nf + 255) / 256, 256, 0, stream>>>(acc, conv_b + l * HDIM, nf);
    k_scatter<<<scatterBlocks, 256, 0, stream>>>(hw, srcA, dstA, dis, acc, ETOT, E);
    k_relu<<<(nf + 255) / 256, 256, 0, stream>>>(acc, hbuf, nf);
  }

  // mean pool
  k_fill<<<(G * HDIM + 255) / 256, 256, 0, stream>>>(pooled, 0.0f, G * HDIM);
  k_fill<<<1, 256, 0, stream>>>(cnt, 0.0f, G);
  k_pool<<<(int)(((size_t)N * 32 + 255) / 256), 256, 0, stream>>>(hbuf, batch, pooled, cnt, N);
  k_pool_div<<<(G * HDIM + 255) / 256, 256, 0, stream>>>(pooled, cnt, G * HDIM);

  // head
  k_lin1<<<(G * HDIM + 255) / 256, 256, 0, stream>>>(pooled, w1, b1, z);
  k_head<<<1, 128, 0, stream>>>(z, w2, b2, out);
}
